// LRU_85538568667436
// MI455X (gfx1250) — matrix operand order repaired and checked by reading, not compiled, not run on hardware
//
#include <hip/hip_runtime.h>
#include <hip/hip_bf16.h>

// ---------------- problem constants (fixed-shape task) ----------------
#define BB      4
#define TT      4096
#define DIN     1024
#define DST     2048
#define BT      (BB * TT)        // 16384
#define CHUNK   64
#define NCHUNK  (TT / CHUNK)     // 64

typedef __attribute__((ext_vector_type(16))) __bf16 v16bf;
typedef __attribute__((ext_vector_type(8)))  __bf16 v8bf;
typedef __attribute__((ext_vector_type(4)))  __bf16 v4bf;
typedef __attribute__((ext_vector_type(8)))  float  v8f;
typedef __attribute__((ext_vector_type(4)))  float  v4f;

// Fragment union: two 16-byte halves land in the adjacent VGPRs of the
// 32-byte WMMA operand (no element shuffling).
union AFrag {
    v16bf v;
    struct { v8bf lo; v8bf hi; } h;
};

// ---------------- fp32 -> bf16 conversion (vector x4) ----------------
__global__ void cvt_f32_bf16_kernel(const float* __restrict__ src,
                                    __bf16* __restrict__ dst, int n4) {
    int i = blockIdx.x * blockDim.x + threadIdx.x;
    if (i < n4) {
        v4f v = *(const v4f*)(src + (size_t)i * 4);
        v4bf o;
        o[0] = (__bf16)v[0]; o[1] = (__bf16)v[1];
        o[2] = (__bf16)v[2]; o[3] = (__bf16)v[3];
        *(v4bf*)(dst + (size_t)i * 4) = o;
    }
}

// ---------------- register-blocked WMMA GEMM ----------------
// C(MxN) = A(MxK) * W(N,K)^T.
// 8 waves/block arranged 2(M) x 4(N); wave tile 64x32 (4x2 WMMA frags);
// block tile 128x128.
//
// A-frag per-lane layout (16-bit A 16x32; k order chosen here, not the ISA table order):
//   lane L<16 : row M=L,    K = k0..k0+15 (contiguous)
//   lane L>=16: row M=L-16, K = k0+16..k0+31
// B-frag (B = W^T, W row-major (N,K) => column n of B is row n of W):
//   lane L<16 : col N=L,    K = k0..k0+15 (contiguous)
//   lane L>=16: col N=L-16, K = k0+16..k0+31
//
// MODE 0: out_bf[m*N+n] = bf16( acc * (1 - sigmoid(log_lambda[n])) )
// MODE 1: out_f [m*N+n] = acc + u[m*N+n] * Dvec[n]
template <int MODE>
__global__ __launch_bounds__(256)
void gemm_wmma_kernel(const __bf16* __restrict__ A,
                      const __bf16* __restrict__ W,
                      int M, int N, int K,
                      __bf16* __restrict__ out_bf,
                      float* __restrict__ out_f,
                      const float* __restrict__ log_lambda,
                      const float* __restrict__ u,
                      const float* __restrict__ Dvec) {
    const int lane = threadIdx.x & 31;
    const int wave = threadIdx.x >> 5;
    const int wm = wave & 1;           // 2 waves along M
    const int wn = wave >> 1;          // 4 waves along N
    const int m0w = blockIdx.x * 128 + wm * 64;   // wave M tile: 64
    const int n0w = blockIdx.y * 128 + wn * 32;   // wave N tile: 32

    const int lr = lane & 15;          // lane row/col within frag
    const int lh = lane >> 4;          // half-select

    // Hoisted per-lane base pointers; K-loop indexes fold into load immediates.
    const __bf16* pa[4];
#pragma unroll
    for (int i = 0; i < 4; ++i)
        pa[i] = A + (size_t)(m0w + i * 16 + lr) * K + (lh << 4);
    const __bf16* pb[2];
#pragma unroll
    for (int j = 0; j < 2; ++j)
        pb[j] = W + (size_t)(n0w + j * 16 + lr) * K + (lh << 4);

    v8f acc[4][2];
#pragma unroll
    for (int i = 0; i < 4; ++i)
#pragma unroll
        for (int j = 0; j < 2; ++j) acc[i][j] = v8f{};

#pragma unroll 2
    for (int k0 = 0; k0 < K; k0 += 32) {
        AFrag a[4];
#pragma unroll
        for (int i = 0; i < 4; ++i) {
            a[i].h.lo = *(const v8bf*)(pa[i] + k0);        // K = kb..kb+7
            a[i].h.hi = *(const v8bf*)(pa[i] + k0 + 8);    // K = kb+8..kb+15
        }
#pragma unroll
        for (int j = 0; j < 2; ++j) {
            v16bf b = *(const v16bf*)(pb[j] + k0);         // 32B contiguous
#pragma unroll
            for (int i = 0; i < 4; ++i)
                acc[i][j] = __builtin_amdgcn_wmma_f32_16x16x32_bf16(
                    false, a[i].v, false, b, (short)0, acc[i][j], false, false);
        }
    }

    // Epilogue. C layout: VGPR r -> row (lane<16 ? r : r+8), col = lane&15.
    const int rb = lh << 3;            // 0 or 8
#pragma unroll
    for (int j = 0; j < 2; ++j) {
        const int col = n0w + j * 16 + lr;
        float scale = 1.0f, dv = 0.0f;
        if (MODE == 0) {
            float lam = 1.0f / (1.0f + expf(-log_lambda[col]));
            scale = 1.0f - lam;
        } else {
            dv = Dvec[col];
        }
#pragma unroll
        for (int i = 0; i < 4; ++i) {
            const int row0 = m0w + i * 16 + rb;
#pragma unroll
            for (int r = 0; r < 8; ++r) {
                size_t idx = (size_t)(row0 + r) * N + col;
                float v = acc[i][j][r];
                if (MODE == 0) out_bf[idx] = (__bf16)(v * scale);
                else           out_f[idx] = v + u[idx] * dv;
            }
        }
    }
}

// ---------------- chunked parallel scan over T ----------------
// Pass A: local chunk sums E_c with zero init
__global__ void scan_ends_kernel(const __bf16* __restrict__ w,
                                 const float* __restrict__ log_lambda,
                                 float* __restrict__ ends) {
    const int s = blockIdx.x * blockDim.x + threadIdx.x;   // state channel
    const int c = blockIdx.y;                              // chunk
    const int b = blockIdx.z;                              // batch
    const float lam = 1.0f / (1.0f + expf(-log_lambda[s]));
    const __bf16* p = w + ((size_t)(b * TT + c * CHUNK) * DST + s);
    float x = 0.0f;
#pragma unroll 8
    for (int i = 0; i < CHUNK; ++i)
        x = fmaf(lam, x, (float)p[(size_t)i * DST]);
    ends[((size_t)b * NCHUNK + c) * DST + s] = x;
}

// Pass B: carry_c = lam^CHUNK * carry_{c-1} + E_{c-1}, carry_0 = 0
__global__ void scan_carry_kernel(const float* __restrict__ ends,
                                  const float* __restrict__ log_lambda,
                                  float* __restrict__ carries) {
    const int id = blockIdx.x * blockDim.x + threadIdx.x;  // 0 .. B*DST-1
    const int b = id / DST, s = id - b * DST;
    const float lam = 1.0f / (1.0f + expf(-log_lambda[s]));
    float lC = lam;
#pragma unroll
    for (int i = 0; i < 6; ++i) lC *= lC;                  // lam^64
    float carry = 0.0f;
    for (int c = 0; c < NCHUNK; ++c) {
        const size_t idx = ((size_t)b * NCHUNK + c) * DST + s;
        carries[idx] = carry;
        carry = fmaf(lC, carry, ends[idx]);
    }
}

// Pass C: re-scan each chunk from its carry, emit xs in bf16
__global__ void scan_apply_kernel(const __bf16* __restrict__ w,
                                  const float* __restrict__ log_lambda,
                                  const float* __restrict__ carries,
                                  __bf16* __restrict__ xs) {
    const int s = blockIdx.x * blockDim.x + threadIdx.x;
    const int c = blockIdx.y;
    const int b = blockIdx.z;
    const float lam = 1.0f / (1.0f + expf(-log_lambda[s]));
    float x = carries[((size_t)b * NCHUNK + c) * DST + s];
    const size_t base = (size_t)(b * TT + c * CHUNK) * DST + s;
    const __bf16* p = w + base;
    __bf16* q = xs + base;
#pragma unroll 8
    for (int i = 0; i < CHUNK; ++i) {
        x = fmaf(lam, x, (float)p[(size_t)i * DST]);
        q[(size_t)i * DST] = (__bf16)x;
    }
}

// ---------------- launcher ----------------
extern "C" void kernel_launch(void* const* d_in, const int* in_sizes, int n_in,
                              void* d_out, int out_size, void* d_ws, size_t ws_size,
                              hipStream_t stream) {
    const float* u_f32    = (const float*)d_in[0];  // (B,T,DIN)
    const float* Win_f32  = (const float*)d_in[1];  // (DST,DIN)
    const float* loglam   = (const float*)d_in[2];  // (DST,)
    const float* Wout_f32 = (const float*)d_in[3];  // (DIN,DST)
    const float* Dvec     = (const float*)d_in[4];  // (DIN,)
    float* y = (float*)d_out;                       // (B,T,DIN) fp32

    // workspace carve-up (bytes)
    char* ws = (char*)d_ws;
    __bf16* u_bf    = (__bf16*)(ws);                        // 32 MB
    __bf16* Win_bf  = (__bf16*)(ws + 33554432);             //  4 MB
    __bf16* Wout_bf = (__bf16*)(ws + 37748736);             //  4 MB
    __bf16* w_proj  = (__bf16*)(ws + 41943040);             // 64 MB
    __bf16* xs      = (__bf16*)(ws + 109051904);            // 64 MB
    float*  ends    = (float* )(ws + 176160768);            //  2 MB
    float*  carries = (float* )(ws + 178257920);            //  2 MB

    // 1) fp32 -> bf16 conversions
    {
        int n4 = (BT * DIN) / 4;                            // 4,194,304
        cvt_f32_bf16_kernel<<<n4 / 256, 256, 0, stream>>>(u_f32, u_bf, n4);
        int w4 = (DST * DIN) / 4;                           //   524,288
        cvt_f32_bf16_kernel<<<w4 / 256, 256, 0, stream>>>(Win_f32, Win_bf, w4);
        cvt_f32_bf16_kernel<<<w4 / 256, 256, 0, stream>>>(Wout_f32, Wout_bf, w4);
    }

    // 2) GEMM1: w_proj = (1-lam) * (u @ Win^T)   [M=BT, N=DST, K=DIN]
    gemm_wmma_kernel<0><<<dim3(BT / 128, DST / 128), 256, 0, stream>>>(
        u_bf, Win_bf, BT, DST, DIN, w_proj, nullptr, loglam, nullptr, nullptr);

    // 3) chunked scan over T
    scan_ends_kernel<<<dim3(DST / 256, NCHUNK, BB), 256, 0, stream>>>(
        w_proj, loglam, ends);
    scan_carry_kernel<<<(BB * DST) / 256, 256, 0, stream>>>(
        ends, loglam, carries);
    scan_apply_kernel<<<dim3(DST / 256, NCHUNK, BB), 256, 0, stream>>>(
        w_proj, loglam, carries, xs);

    // 4) GEMM2: y = xs @ Wout^T + u * D          [M=BT, N=DIN, K=DST]
    gemm_wmma_kernel<1><<<dim3(BT / 128, DIN / 128), 256, 0, stream>>>(
        xs, Wout_bf, BT, DIN, DST, nullptr, y, nullptr, u_f32, Dvec);

    (void)in_sizes; (void)n_in; (void)out_size; (void)ws_size;
}
